// GroupedQueryAttention_72954314489857
// MI455X (gfx1250) — compile-verified, matrix operand order checked by reading, not run on hardware
//
#include <hip/hip_runtime.h>

// ---------------------------------------------------------------------------
// GQA block for MI455X (gfx1250, wave32, WMMA, async LDS copies).
// All matmuls: v_wmma_f32_16x16x32_bf16 (bf16 in, f32 accumulate).
// f32->bf16 conversion + weight transposition done ONCE up front so the
// GEMM / attention hot loops move pure 16B bf16 chunks (async-to-LDS capable).
// ---------------------------------------------------------------------------

typedef __bf16 bf16_t;
typedef __attribute__((ext_vector_type(16))) __bf16 v16bf;
typedef __attribute__((ext_vector_type(8)))  __bf16 v8bf;
typedef __attribute__((ext_vector_type(8)))  float   v8f;
typedef __attribute__((ext_vector_type(4)))  float   v4f;
typedef int v4i __attribute__((vector_size(16)));   // matches builtin param type

static __device__ __forceinline__ bf16_t f2bf(float f) { return (bf16_t)f; }

static __device__ __forceinline__ v16bf make16(v8bf lo, v8bf hi) {
  v16bf r;
#pragma unroll
  for (int i = 0; i < 8; ++i) { r[i] = lo[i]; r[i + 8] = hi[i]; }
  return r;
}

static __device__ __forceinline__ v8f wmma_bf16(v16bf a, v16bf b, v8f c) {
  return __builtin_amdgcn_wmma_f32_16x16x32_bf16(false, a, false, b,
                                                 (short)0, c, false, false);
}

// ---- async 16B global->LDS copy (gfx1250 GLOBAL_LOAD_ASYNC_TO_LDS_B128) ----
#if defined(__has_builtin)
#if __has_builtin(__builtin_amdgcn_global_load_async_to_lds_b128) && \
    __has_builtin(__builtin_amdgcn_s_wait_asynccnt)
#define HAVE_ASYNC_LDS 1
#endif
#endif

static __device__ __forceinline__ void async_copy16(const bf16_t* g, bf16_t* s) {
#ifdef HAVE_ASYNC_LDS
  // AS1/AS3 pointers built via integer casts: low 32 bits of a flat LDS
  // address are the LDS byte offset (ISA 10.2 aperture mapping).
  __attribute__((address_space(1))) v4i* gp =
      (__attribute__((address_space(1))) v4i*)(unsigned long long)(size_t)g;
  __attribute__((address_space(3))) v4i* lp =
      (__attribute__((address_space(3))) v4i*)(unsigned)(size_t)s;
  __builtin_amdgcn_global_load_async_to_lds_b128(gp, lp, 0, 0);
#else
  *(v8bf*)s = *(const v8bf*)g;
#endif
}
static __device__ __forceinline__ void wait_async0() {
#ifdef HAVE_ASYNC_LDS
  __builtin_amdgcn_s_wait_asynccnt(0);
#endif
}
static __device__ __forceinline__ void wait_async8() {
#ifdef HAVE_ASYNC_LDS
  __builtin_amdgcn_s_wait_asynccnt(8);
#endif
}

// ---------------------------------------------------------------------------
// One-time converters (out of the hot loops).
// ---------------------------------------------------------------------------
__global__ __launch_bounds__(256)
void convert_f32_to_bf16_kernel(const float* __restrict__ in,
                                bf16_t* __restrict__ out, int n8)
{
  int i = blockIdx.x * 256 + threadIdx.x;
  if (i >= n8) return;
  v4f a = ((const v4f*)in)[2 * i];
  v4f b = ((const v4f*)in)[2 * i + 1];
  v8bf o;
  o[0] = f2bf(a.x); o[1] = f2bf(a.y); o[2] = f2bf(a.z); o[3] = f2bf(a.w);
  o[4] = f2bf(b.x); o[5] = f2bf(b.y); o[6] = f2bf(b.z); o[7] = f2bf(b.w);
  ((v8bf*)out)[i] = o;
}

// in[R][C] f32  ->  out[C][R] bf16   (32x32 LDS tile transpose)
__global__ __launch_bounds__(256)
void transpose_f32_to_bf16_kernel(const float* __restrict__ in,
                                  bf16_t* __restrict__ out, int R, int C)
{
  __shared__ bf16_t tile[32][33];
  const int tid = threadIdx.x;
  const int r0 = blockIdx.y * 32, c0 = blockIdx.x * 32;
#pragma unroll
  for (int i = 0; i < 4; ++i) {
    int rr = (tid >> 5) + i * 8, cc = tid & 31;
    tile[rr][cc] = f2bf(in[(size_t)(r0 + rr) * C + c0 + cc]);
  }
  __syncthreads();
#pragma unroll
  for (int i = 0; i < 4; ++i) {
    int rr = (tid >> 5) + i * 8, cc = tid & 31;
    out[(size_t)(c0 + rr) * R + r0 + cc] = tile[cc][rr];
  }
}

// ---------------------------------------------------------------------------
// TN GEMM: C[M,N] = A[M,K] * Bt[N,K]^T, bf16 operands, f32 accumulate.
// 128x128 block tile, 64-deep K stage, double-buffered async LDS copies.
// 8 waves; wave owns 32x64 (2x4 16x16 accumulators).
// ---------------------------------------------------------------------------
__global__ __launch_bounds__(256)
void gemm_bf16_tn_kernel(const bf16_t* __restrict__ A,   // [M][K]
                         const bf16_t* __restrict__ Bt,  // [N][K]
                         float*  __restrict__ Cf,        // f32 out (or null)
                         bf16_t* __restrict__ Cb,        // bf16 out (or null)
                         int M, int N, int K)
{
  __shared__ __align__(16) bf16_t As[2][128 * 64];
  __shared__ __align__(16) bf16_t Bs[2][128 * 64];

  const int tid   = threadIdx.x;
  const int wave  = tid >> 5;
  const int lane  = tid & 31;
  const int lhalf = lane >> 4;
  const int l16   = lane & 15;
  const int waveM = wave >> 1;
  const int waveN = wave & 1;
  const int m0 = blockIdx.x * 128;
  const int n0 = blockIdx.y * 128;

  const bf16_t* gA = A  + (size_t)m0 * K;
  const bf16_t* gB = Bt + (size_t)n0 * K;

  const v8f vzero = {0.f, 0.f, 0.f, 0.f, 0.f, 0.f, 0.f, 0.f};
  v8f acc[2][4];
#pragma unroll
  for (int mt = 0; mt < 2; ++mt)
#pragma unroll
    for (int nt = 0; nt < 4; ++nt) acc[mt][nt] = vzero;

  auto issue = [&](int st) {
    const bf16_t* a = gA + (st << 6);
    const bf16_t* b = gB + (st << 6);
    bf16_t* sa = As[st & 1];
    bf16_t* sb = Bs[st & 1];
#pragma unroll
    for (int j = 0; j < 4; ++j) {          // 1024 16B chunks / 256 threads
      int c = tid + 256 * j;
      int row = c >> 3, ko = (c & 7) << 3; // 8 chunks per 64-elem row
      async_copy16(a + (size_t)row * K + ko, sa + row * 64 + ko);
      async_copy16(b + (size_t)row * K + ko, sb + row * 64 + ko);
    }
  };

  const int nst = K >> 6;
  issue(0);
  if (nst > 1) issue(1);

  for (int st = 0; st < nst; ++st) {
    if (st + 1 < nst) wait_async8(); else wait_async0();
    __syncthreads();
    const int bsel = st & 1;
#pragma unroll
    for (int kk = 0; kk < 64; kk += 32) {
      v16bf af[2], bfr[4];
#pragma unroll
      for (int mt = 0; mt < 2; ++mt) {
        int row = waveM * 32 + mt * 16 + l16;
        v8bf lo = *(const v8bf*)&As[bsel][row * 64 + kk + lhalf * 8];
        v8bf hi = *(const v8bf*)&As[bsel][row * 64 + kk + 16 + lhalf * 8];
        af[mt] = make16(lo, hi);
      }
#pragma unroll
      for (int nt = 0; nt < 4; ++nt) {
        int col = waveN * 64 + nt * 16 + l16;
        v8bf lo = *(const v8bf*)&Bs[bsel][col * 64 + kk + lhalf * 8];
        v8bf hi = *(const v8bf*)&Bs[bsel][col * 64 + kk + 16 + lhalf * 8];
        bfr[nt] = make16(lo, hi);
      }
#pragma unroll
      for (int mt = 0; mt < 2; ++mt)
#pragma unroll
        for (int nt = 0; nt < 4; ++nt)
          acc[mt][nt] = wmma_bf16(af[mt], bfr[nt], acc[mt][nt]);
    }
    __syncthreads();
    if (st + 2 < nst) issue(st + 2);
  }

#pragma unroll
  for (int mt = 0; mt < 2; ++mt)
#pragma unroll
    for (int nt = 0; nt < 4; ++nt)
#pragma unroll
      for (int r = 0; r < 8; ++r) {
        int row = m0 + waveM * 32 + mt * 16 + r + 8 * lhalf;
        int col = n0 + waveN * 64 + nt * 16 + l16;
        if (Cb) Cb[(size_t)row * N + col] = f2bf(acc[mt][nt][r]);
        else    Cf[(size_t)row * N + col] = acc[mt][nt][r];
      }
}

// ---------------------------------------------------------------------------
// RoPE in place on bf16 (B,S,H,128); optional output scale (1/sqrt(HD) for Q).
// ---------------------------------------------------------------------------
__global__ __launch_bounds__(256)
void rope_bf16_kernel(bf16_t* __restrict__ t, const float* __restrict__ cosb,
                      const float* __restrict__ sinb, int S, int H,
                      float scale, int total)
{
  int i = blockIdx.x * 256 + threadIdx.x;
  if (i >= total) return;
  int d2  = i & 63;           // HD/2 == 64
  int rem = i >> 6;           // (b*S + s)*H + h
  int h   = rem % H;
  int bs  = rem / H;
  int s   = bs % S;
  float c  = cosb[s * 64 + d2];
  float sn = sinb[s * 64 + d2];
  bf16_t* p = t + ((size_t)bs * H + h) * 128 + 2 * d2;
  float a = (float)p[0], b = (float)p[1];
  p[0] = f2bf((a * c - b * sn) * scale);
  p[1] = f2bf((a * sn + b * c) * scale);
}

// ---------------------------------------------------------------------------
// Causal GQA flash attention, bf16 in / bf16 out (accumulate f32).
// Block = (b, h, 128 q rows); wave w owns rows [q0+16w, q0+16w+15].
// 32-key tiles: K tile async-copied to LDS, V transposed at store.
// ---------------------------------------------------------------------------
__global__ __launch_bounds__(256)
void gqa_flash_kernel(const bf16_t* __restrict__ q,   // (B,S,32,128), pre-scaled
                      const bf16_t* __restrict__ k,   // (B,S,8,128)
                      const bf16_t* __restrict__ v,   // (B,S,8,128)
                      bf16_t* __restrict__ o,         // (B*S, 4096)
                      int S)
{
  constexpr int HQc = 32, HKVc = 8, HDc = 128, NREPc = 4;
  __shared__ __align__(16) bf16_t Ks[32 * 128];   // [key][hd]
  __shared__ __align__(16) bf16_t Vs[128 * 32];   // [hd][key]
  __shared__ __align__(16) bf16_t Ps[8 * 16 * 32];// per-wave [row][key]

  const int tid   = threadIdx.x;
  const int wave  = tid >> 5;
  const int lane  = tid & 31;
  const int lhalf = lane >> 4;
  const int l16   = lane & 15;

  const int bh = blockIdx.y;
  const int b  = bh / HQc, h = bh % HQc, hk = h / NREPc;
  const int q0 = blockIdx.x * 128;

  // Q A-fragments, loaded once (already scaled by 1/sqrt(HD) in rope pass)
  const int qrow = q0 + wave * 16 + l16;
  const bf16_t* qp = q + (((size_t)b * S + qrow) * HQc + h) * HDc;
  v16bf aq[4];
#pragma unroll
  for (int ks = 0; ks < 4; ++ks) {
    v8bf lo = *(const v8bf*)(qp + ks * 32 + lhalf * 8);
    v8bf hi = *(const v8bf*)(qp + ks * 32 + 16 + lhalf * 8);
    aq[ks] = make16(lo, hi);
  }

  const v8f vzero = {0.f, 0.f, 0.f, 0.f, 0.f, 0.f, 0.f, 0.f};
  v8f oacc[8];
  float mrow[8], lrow[8];
#pragma unroll
  for (int i = 0; i < 8; ++i) { oacc[i] = vzero; mrow[i] = -1e30f; lrow[i] = 0.f; }

  const int khi_wave = q0 + wave * 16 + 15;
  const int nkt = (q0 + 128) >> 5;

  for (int kt = 0; kt < nkt; ++kt) {
    const int kb = kt * 32;
    // K tile: 512 16B chunks, async; V tile: load + transposed scatter
#pragma unroll
    for (int j = 0; j < 2; ++j) {
      int c = tid + 256 * j;
      int key = c >> 4, ho = (c & 15) << 3;    // 16 chunks per 128-elem row
      const bf16_t* kp = k + (((size_t)b * S + kb + key) * HKVc + hk) * HDc + ho;
      async_copy16(kp, &Ks[key * 128 + ho]);
      const bf16_t* vp = v + (((size_t)b * S + kb + key) * HKVc + hk) * HDc + ho;
      v8bf vv = *(const v8bf*)vp;
#pragma unroll
      for (int e = 0; e < 8; ++e) Vs[(ho + e) * 32 + key] = vv[e];
    }
    wait_async0();
    __syncthreads();

    if (kb <= khi_wave) {
      // S = Q K^T (16 x 32), 4 K-steps over HD
      v8f sacc[2] = {vzero, vzero};
#pragma unroll
      for (int ks = 0; ks < 4; ++ks)
#pragma unroll
        for (int nt = 0; nt < 2; ++nt) {
          int key = nt * 16 + l16;
          v8bf lo = *(const v8bf*)&Ks[key * 128 + ks * 32 + lhalf * 8];
          v8bf hi = *(const v8bf*)&Ks[key * 128 + ks * 32 + 16 + lhalf * 8];
          sacc[nt] = wmma_bf16(aq[ks], make16(lo, hi), sacc[nt]);
        }

      // causal mask + online softmax (row r + 8*lhalf lives in one 16-lane half)
      float pv0[8], pv1[8];
#pragma unroll
      for (int r = 0; r < 8; ++r) {
        int rowg = q0 + wave * 16 + r + 8 * lhalf;
        float s0 = sacc[0][r]; if (kb + l16      > rowg) s0 = -1e30f;
        float s1 = sacc[1][r]; if (kb + 16 + l16 > rowg) s1 = -1e30f;
        float mx = fmaxf(s0, s1);
#pragma unroll
        for (int msk = 1; msk <= 8; msk <<= 1)
          mx = fmaxf(mx, __shfl_xor(mx, msk, 32));
        float mnew = fmaxf(mrow[r], mx);
        float scl  = __expf(mrow[r] - mnew);
        float p0 = __expf(s0 - mnew);
        float p1 = __expf(s1 - mnew);
        float ps = p0 + p1;
#pragma unroll
        for (int msk = 1; msk <= 8; msk <<= 1)
          ps += __shfl_xor(ps, msk, 32);
        lrow[r] = lrow[r] * scl + ps;
        mrow[r] = mnew;
#pragma unroll
        for (int ntd = 0; ntd < 8; ++ntd) oacc[ntd][r] *= scl;
        pv0[r] = p0; pv1[r] = p1;
      }

      // C-layout -> A-layout for P via wave-private LDS (same-wave DS order)
#pragma unroll
      for (int r = 0; r < 8; ++r) {
        int rowl = r + 8 * lhalf;
        Ps[wave * 512 + rowl * 32 + l16]      = f2bf(pv0[r]);
        Ps[wave * 512 + rowl * 32 + 16 + l16] = f2bf(pv1[r]);
      }
      v8bf plo = *(const v8bf*)&Ps[wave * 512 + l16 * 32 + lhalf * 8];
      v8bf phi = *(const v8bf*)&Ps[wave * 512 + l16 * 32 + 16 + lhalf * 8];
      v16bf pa = make16(plo, phi);

      // O += P V
#pragma unroll
      for (int ntd = 0; ntd < 8; ++ntd) {
        int hd = ntd * 16 + l16;
        v8bf lo = *(const v8bf*)&Vs[hd * 32 + lhalf * 8];
        v8bf hi = *(const v8bf*)&Vs[hd * 32 + 16 + lhalf * 8];
        oacc[ntd] = wmma_bf16(pa, make16(lo, hi), oacc[ntd]);
      }
    }
    __syncthreads();
  }

  // normalize, emit bf16 for the final GEMM
#pragma unroll
  for (int ntd = 0; ntd < 8; ++ntd)
#pragma unroll
    for (int r = 0; r < 8; ++r) {
      int rowg = q0 + wave * 16 + r + 8 * lhalf;
      float val = oacc[ntd][r] / lrow[r];
      o[((size_t)b * S + rowg) * (HQc * HDc) + h * HDc + ntd * 16 + l16] = f2bf(val);
    }
}

// ---------------------------------------------------------------------------
// Launch: convert/transpose -> QKV GEMMs -> RoPE -> attention -> out GEMM.
// ---------------------------------------------------------------------------
extern "C" void kernel_launch(void* const* d_in, const int* in_sizes, int n_in,
                              void* d_out, int out_size, void* d_ws, size_t ws_size,
                              hipStream_t stream) {
  (void)in_sizes; (void)n_in; (void)out_size; (void)ws_size;
  const float* x    = (const float*)d_in[0];
  const float* w_q  = (const float*)d_in[1];
  const float* w_k  = (const float*)d_in[2];
  const float* w_v  = (const float*)d_in[3];
  const float* w_o  = (const float*)d_in[4];
  const float* fcos = (const float*)d_in[7];
  const float* fsin = (const float*)d_in[8];
  float* out = (float*)d_out;

  const int B = 2, S = 2048, D = 4096, HQ = 32, HKV = 8, HD = 128;
  const int M = B * S;                           // 4096
  const size_t nq  = (size_t)B * S * HQ * HD;    // 16,777,216
  const size_t nkv = (size_t)B * S * HKV * HD;   //  4,194,304

  bf16_t* w = (bf16_t*)d_ws;
  bf16_t* wqT = w;            w += nq;    // [4096][4096]
  bf16_t* wkT = w;            w += nkv;   // [1024][4096]
  bf16_t* wvT = w;            w += nkv;   // [1024][4096]
  bf16_t* woT = w;            w += nq;    // [4096][4096]
  bf16_t* xb  = w;            w += nq;    // x bf16; later reused as attn out
  bf16_t* xqb = w;            w += nq;
  bf16_t* xkb = w;            w += nkv;
  bf16_t* xvb = w;            w += nkv;
  bf16_t* aob = xb;                        // alias: x dead after QKV GEMMs

  dim3 blk(256);

  // one-time conversions / transpositions
  convert_f32_to_bf16_kernel<<<(int)(nq / 8 / 256), blk, 0, stream>>>(x, xb, (int)(nq / 8));
  transpose_f32_to_bf16_kernel<<<dim3(D / 32, D / 32), blk, 0, stream>>>(w_q, wqT, D, D);
  transpose_f32_to_bf16_kernel<<<dim3((HKV * HD) / 32, D / 32), blk, 0, stream>>>(w_k, wkT, D, HKV * HD);
  transpose_f32_to_bf16_kernel<<<dim3((HKV * HD) / 32, D / 32), blk, 0, stream>>>(w_v, wvT, D, HKV * HD);
  transpose_f32_to_bf16_kernel<<<dim3(D / 32, D / 32), blk, 0, stream>>>(w_o, woT, D, D);

  // QKV projections (bf16 out)
  gemm_bf16_tn_kernel<<<dim3(M / 128, (HQ * HD) / 128), blk, 0, stream>>>(
      xb, wqT, nullptr, xqb, M, HQ * HD, D);
  gemm_bf16_tn_kernel<<<dim3(M / 128, (HKV * HD) / 128), blk, 0, stream>>>(
      xb, wkT, nullptr, xkb, M, HKV * HD, D);
  gemm_bf16_tn_kernel<<<dim3(M / 128, (HKV * HD) / 128), blk, 0, stream>>>(
      xb, wvT, nullptr, xvb, M, HKV * HD, D);

  // RoPE (fold 1/sqrt(HD) into Q)
  const int nprq = B * S * HQ * (HD / 2);
  rope_bf16_kernel<<<(nprq + 255) / 256, blk, 0, stream>>>(
      xqb, fcos, fsin, S, HQ, 0.08838834764831845f, nprq);
  const int nprk = B * S * HKV * (HD / 2);
  rope_bf16_kernel<<<(nprk + 255) / 256, blk, 0, stream>>>(
      xkb, fcos, fsin, S, HKV, 1.0f, nprk);

  // causal GQA flash attention (bf16 out -> aob)
  gqa_flash_kernel<<<dim3(S / 128, B * HQ), blk, 0, stream>>>(xqb, xkb, xvb, aob, S);

  // output projection (f32 out)
  gemm_bf16_tn_kernel<<<dim3(M / 128, D / 128), blk, 0, stream>>>(
      aob, woT, out, nullptr, M, D, D);
}
